// Conv2dVQModel_72602127172079
// MI455X (gfx1250) — compile-verified, matrix operand order checked by reading, not run on hardware
//
#include <hip/hip_runtime.h>
#include <hip/hip_bf16.h>

typedef _Float16 h16;
typedef __attribute__((ext_vector_type(8)))  _Float16 v8h;
typedef __attribute__((ext_vector_type(16))) _Float16 v16h;
typedef __attribute__((ext_vector_type(8)))  float    v8f;
typedef __attribute__((ext_vector_type(4)))  float    v4f;

// ---------------------------------------------------------------------------
// Generic fused conv-as-GEMM with WMMA f16->f32.
//   out[b][m][t] = act( alpha[m] * sum_{kw,cr} W[m][kw*CRP+cr]*in[b][cr][t+kw-PAD] + beta[m] )
// Block: 256 thr = 8 waves; tile 32 couts x 64 positions (wave = 16x16 tile).
// LDS holds input tile as f16 [t_local][cr] (cr contiguous -> b128 B-fragment loads).
// Weights pre-packed f16, k = kw*CRP + cr (cr = cin*KH + kh), rows padded to MPAD.
// Intermediates are stored f16 (halves HBM traffic; working set fits 192MB L2);
// TIN/TOUT select f32 at the graph edges (x, x_recon, states).
// ---------------------------------------------------------------------------
template <typename TIN, typename TOUT,
          int CR, int CRP, int KW, int PAD, int COUT, bool RELU>
__global__ __launch_bounds__(256)
void conv_wmma(const TIN* __restrict__ in, TOUT* __restrict__ out,
               const h16* __restrict__ wp, const float* __restrict__ alpha,
               const float* __restrict__ beta, int T,
               long in_bstride, long out_bstride, long out_cstride)
{
  constexpr int K  = KW * CRP;          // multiple of 32
  constexpr int BN = 64;
  constexpr int NT = BN + KW - 1;
  __shared__ __align__(16) h16 raw[NT * CRP];

  const int b   = blockIdx.z;
  const int t0  = blockIdx.x * BN;
  const int tid = threadIdx.x;
  const TIN* inb = in + (long)b * in_bstride;

  // stage input tile (t-minor for coalesced global reads)
  for (int e = tid; e < NT * CRP; e += 256) {
    int cr = e / NT;
    int tl = e - cr * NT;
    int t  = t0 + tl - PAD;
    float v = 0.f;
    if (cr < CR && t >= 0 && t < T) v = (float)inb[(long)cr * T + t];
    raw[tl * CRP + cr] = (h16)v;
  }
  __syncthreads();

  const int w    = tid >> 5;
  const int lane = tid & 31;
  const int ln   = lane & 15;
  const int hi   = lane >> 4;
  const int mbase = blockIdx.y * 32 + (w >> 2) * 16;
  const int n     = (w & 3) * 16 + ln;      // local position in tile
  const int t     = t0 + n;

  v8f acc = {};
  const h16* wrow = wp + (long)(mbase + ln) * K;

#pragma unroll
  for (int kk = 0; kk < K / 32; ++kk) {
    const int koff = kk * 32 + hi * 8;
    // A fragment: weights (global f16, 2 x 16B)
    v8h alo = *reinterpret_cast<const v8h*>(wrow + koff);
    v8h ahi = *reinterpret_cast<const v8h*>(wrow + koff + 16);
    if (kk + 1 < K / 32) __builtin_prefetch(wrow + koff + 32, 0, 3);
    v16h A;
#pragma unroll
    for (int i = 0; i < 8; ++i) { A[i] = alo[i]; A[8 + i] = ahi[i]; }
    // B fragment: LDS (2 x 16B, contiguous cr run inside one kw block)
    const int kw  = koff / CRP;
    const int cr0 = koff & (CRP - 1);
    const h16* bp = &raw[(n + kw) * CRP + cr0];
    v8h blo = *reinterpret_cast<const v8h*>(bp);
    v8h bhi = *reinterpret_cast<const v8h*>(bp + 16);
    v16h Bm;
#pragma unroll
    for (int i = 0; i < 8; ++i) { Bm[i] = blo[i]; Bm[8 + i] = bhi[i]; }
    acc = __builtin_amdgcn_wmma_f32_16x16x32_f16(false, A, false, Bm,
                                                 (short)0, acc, false, false);
  }

  // fused alpha/beta (+ReLU) epilogue; C layout: lane n, vgpr r -> m = hi*8 + r
  const int mrow = mbase + hi * 8;
  v4f a0 = *reinterpret_cast<const v4f*>(alpha + mrow);
  v4f a1 = *reinterpret_cast<const v4f*>(alpha + mrow + 4);
  v4f b0 = *reinterpret_cast<const v4f*>(beta + mrow);
  v4f b1 = *reinterpret_cast<const v4f*>(beta + mrow + 4);
  if (t < T) {
    TOUT* ob = out + (long)b * out_bstride + t;
#pragma unroll
    for (int r = 0; r < 8; ++r) {
      int m = mrow + r;
      if (m < COUT) {
        float al = (r < 4) ? a0[r] : a1[r - 4];
        float be = (r < 4) ? b0[r] : b1[r - 4];
        float v  = acc[r] * al + be;
        if (RELU) v = v > 0.f ? v : 0.f;
        ob[(long)m * out_cstride] = (TOUT)v;
      }
    }
  }
}

// ---------------------------------------------------------------------------
// VQ argmin via WMMA: d(n,m) = |c_m|^2 - 2 * z_n . c_m  (|z|^2 const per column)
// Block: 128 positions, 8 waves; each wave scans all 256 codes (16 m-tiles).
// ---------------------------------------------------------------------------
__global__ __launch_bounds__(256)
void vq_argmin(const h16* __restrict__ ze, const h16* __restrict__ cb,
               const float* __restrict__ cnorm, int* __restrict__ idxout, int T)
{
  constexpr int BN = 128;
  __shared__ __align__(16) h16 raw[BN * 64];
  const int b   = blockIdx.z;
  const int t0  = blockIdx.x * BN;
  const int tid = threadIdx.x;
  const h16* zb = ze + (long)b * 64 * T;
  for (int e = tid; e < BN * 64; e += 256) {
    int tl = e & 127;
    int d  = e >> 7;
    int t  = t0 + tl;
    raw[tl * 64 + d] = (t < T) ? zb[(long)d * T + t] : (h16)0.f;
  }
  __syncthreads();

  const int w = tid >> 5, lane = tid & 31, ln = lane & 15, hi = lane >> 4;
  const int n = w * 16 + ln;
  const int t = t0 + n;
  float bestd = 3.4e38f; int besti = 0;

  for (int mt = 0; mt < 16; ++mt) {
    v8f acc = {};
    const h16* crow = cb + (mt * 16 + ln) * 64;
#pragma unroll
    for (int kk = 0; kk < 2; ++kk) {
      const int koff = kk * 32 + hi * 8;
      v8h alo = *reinterpret_cast<const v8h*>(crow + koff);
      v8h ahi = *reinterpret_cast<const v8h*>(crow + koff + 16);
      v16h A;
#pragma unroll
      for (int i = 0; i < 8; ++i) { A[i] = alo[i]; A[8 + i] = ahi[i]; }
      const h16* bp = &raw[n * 64 + koff];
      v8h blo = *reinterpret_cast<const v8h*>(bp);
      v8h bhi = *reinterpret_cast<const v8h*>(bp + 16);
      v16h Bm;
#pragma unroll
      for (int i = 0; i < 8; ++i) { Bm[i] = blo[i]; Bm[8 + i] = bhi[i]; }
      acc = __builtin_amdgcn_wmma_f32_16x16x32_f16(false, A, false, Bm,
                                                   (short)0, acc, false, false);
    }
    const int mrow = mt * 16 + hi * 8;
    v4f c0 = *reinterpret_cast<const v4f*>(cnorm + mrow);
    v4f c1 = *reinterpret_cast<const v4f*>(cnorm + mrow + 4);
#pragma unroll
    for (int r = 0; r < 8; ++r) {
      float cn = (r < 4) ? c0[r] : c1[r - 4];
      float d  = cn - 2.f * acc[r];
      if (d < bestd) { bestd = d; besti = mrow + r; }
    }
  }
  // combine lane n (hi=0) with lane n+16 (hi=1); prefer lower index on ties
  float od = __shfl_xor(bestd, 16, 32);
  int   oi = __shfl_xor(besti, 16, 32);
  if (od < bestd || (od == bestd && oi < besti)) { bestd = od; besti = oi; }
  if (hi == 0 && t < T) idxout[(long)b * T + t] = besti;
}

// gather z_q (f16), emit indices-as-float, accumulate commit-loss sum + counts
__global__ __launch_bounds__(256)
void vq_post(const int* __restrict__ idx, const float* __restrict__ cb,
             const h16* __restrict__ ze, h16* __restrict__ zq,
             float* __restrict__ idxf, float* eacc, float* counts, int T)
{
  const int gid = blockIdx.x * 256 + threadIdx.x;
  const bool valid = gid < 128 * T;
  float s = 0.f;
  if (valid) {
    int b = gid / T, t = gid - b * T;
    int id = idx[gid];
    const float* crow = cb + id * 64;
    const h16* zbb = ze + (long)b * 64 * T + t;
    h16* qb = zq + (long)b * 64 * T + t;
    for (int d = 0; d < 64; ++d) {
      float c  = crow[d];
      float df = c - (float)zbb[(long)d * T];
      s += df * df;
      qb[(long)d * T] = (h16)c;
    }
    idxf[gid] = (float)id;
    atomicAdd(counts + id, 1.f);
  }
  __shared__ float red[256];
  red[threadIdx.x] = s;
  __syncthreads();
  for (int st = 128; st > 0; st >>= 1) {
    if (threadIdx.x < st) red[threadIdx.x] += red[threadIdx.x + st];
    __syncthreads();
  }
  if (threadIdx.x == 0) atomicAdd(eacc, red[0]);
}

__global__ void zero_small(float* counts, float* eacc)
{
  counts[threadIdx.x] = 0.f;
  if (threadIdx.x == 0) eacc[0] = 0.f;
}

__global__ void vq_finalize(const float* counts, const float* eacc, float* out)
{
  __shared__ float sp[256], su[256];
  int tid = threadIdx.x;
  float avg = counts[tid] / 128000.f;
  sp[tid] = avg * logf(avg + 1e-10f);
  su[tid] = (avg > 0.f) ? 1.f : 0.f;
  __syncthreads();
  for (int st = 128; st > 0; st >>= 1) {
    if (tid < st) { sp[tid] += sp[tid + st]; su[tid] += su[tid + st]; }
    __syncthreads();
  }
  if (tid == 0) {
    float e = eacc[0] / (128000.f * 64.f);
    out[2305536] = 0.4f * e;     // vq_loss
    out[2305537] = e;            // commit_loss
    out[2305538] = expf(-sp[0]); // perplexity
    out[2305539] = su[0] / 256.f;// usage
  }
}

// mean-pool over T + 2-layer MLP classifier, one block per batch row
__global__ __launch_bounds__(256)
void cls_kernel(const h16* __restrict__ zq, const float* __restrict__ w1,
                const float* __restrict__ b1, const float* __restrict__ w2,
                const float* __restrict__ b2, float* __restrict__ out, int T)
{
  __shared__ float part[256];
  __shared__ float pooled[64];
  __shared__ float hbuf[128];
  const int b = blockIdx.x, tid = threadIdx.x;
  const int c = tid & 63, g = tid >> 6;
  const h16* zc = zq + ((long)b * 64 + c) * T;
  float s = 0.f;
  for (int t = g; t < T; t += 4) s += (float)zc[t];
  part[tid] = s;
  __syncthreads();
  if (tid < 64)
    pooled[tid] = (part[tid] + part[tid + 64] + part[tid + 128] + part[tid + 192]) / (float)T;
  __syncthreads();
  if (tid < 128) {
    float a = b1[tid];
    for (int d = 0; d < 64; ++d) a += pooled[d] * w1[tid * 64 + d];
    hbuf[tid] = a > 0.f ? a : 0.f;
  }
  __syncthreads();
  if (tid < 12) {
    float a = b2[tid];
    for (int j = 0; j < 128; ++j) a += hbuf[j] * w2[tid * 128 + j];
    out[(long)b * 12 + tid] = a;
  }
}

// ---------------- weight / scale preparation ----------------
// pack OIHW f32 -> f16 [MPAD][K], k = kw*CRP + (cin*KH + kh), zero-padded
__global__ void prep_w(const float* __restrict__ w, h16* __restrict__ dst,
                       int Cout, int Cin, int KH, int KWk, int CRP, int K, int MPAD)
{
  long total = (long)MPAD * K;
  for (long i = (long)blockIdx.x * blockDim.x + threadIdx.x; i < total;
       i += (long)gridDim.x * blockDim.x) {
    int m = (int)(i / K), k = (int)(i - (long)m * K);
    int kw = k / CRP, cr = k - kw * CRP;
    float v = 0.f;
    if (m < Cout && cr < Cin * KH) {
      int cin = cr / KH, kh = cr - cin * KH;
      v = w[(((long)m * Cin + cin) * KH + kh) * KWk + kw];
    }
    dst[i] = (h16)v;
  }
}

// ConvTranspose(2,5) weight (in=64,out=9,2,5): flip+transpose, one pack per out-H row
__global__ void prep_wt(const float* __restrict__ w, h16* __restrict__ d0,
                        h16* __restrict__ d1)
{
  int i = blockIdx.x * 256 + threadIdx.x;   // 32 rows x K=320
  if (i >= 32 * 320) return;
  int m = i / 320, k = i - m * 320;
  int kw = k / 64, ci = k - kw * 64;
  float v0 = 0.f, v1 = 0.f;
  if (m < 9) {
    v0 = w[((ci * 9 + m) * 2 + 1) * 5 + (4 - kw)];  // h_out=0 uses kh=1
    v1 = w[((ci * 9 + m) * 2 + 0) * 5 + (4 - kw)];  // h_out=1 uses kh=0
  }
  d0[i] = (h16)v0; d1[i] = (h16)v1;
}

// fold conv-bias (+ optional BN) into per-channel alpha/beta, padded to MPAD
__global__ void prep_ab(const float* bias, const float* g, const float* be,
                        const float* mean, const float* var,
                        float* alpha, float* beta, int C, int MPAD)
{
  int i = blockIdx.x * blockDim.x + threadIdx.x;
  if (i >= MPAD) return;
  float a = 0.f, bt = 0.f;
  if (i < C) {
    if (g) {
      float s = g[i] * rsqrtf(var[i] + 1e-5f);
      a = s; bt = (bias[i] - mean[i]) * s + be[i];
    } else { a = 1.f; bt = bias[i]; }
  }
  alpha[i] = a; beta[i] = bt;
}

__global__ void prep_cb(const float* __restrict__ cb, h16* __restrict__ cbh,
                        float* __restrict__ cnorm)
{
  int i = blockIdx.x * 256 + threadIdx.x;
  if (i < 256 * 64) cbh[i] = (h16)cb[i];
  if (i < 256) {
    float s = 0.f;
    for (int d = 0; d < 64; ++d) { float v = cb[i * 64 + d]; s += v * v; }
    cnorm[i] = s;
  }
}

// ---------------------------------------------------------------------------
extern "C" void kernel_launch(void* const* d_in, const int* in_sizes, int n_in,
                              void* d_out, int out_size, void* d_ws, size_t ws_size,
                              hipStream_t stream)
{
  (void)in_sizes; (void)n_in; (void)out_size; (void)ws_size;
  const int B = 128, T = 1000;

  const float* x      = (const float*)d_in[0];
  const float* enc_w0 = (const float*)d_in[1];  const float* enc_b0 = (const float*)d_in[2];
  const float* enc_w1 = (const float*)d_in[3];  const float* enc_b1 = (const float*)d_in[4];
  const float* e1g = (const float*)d_in[5];  const float* e1b = (const float*)d_in[6];
  const float* e1m = (const float*)d_in[7];  const float* e1v = (const float*)d_in[8];
  const float* enc_w2 = (const float*)d_in[9];  const float* enc_b2 = (const float*)d_in[10];
  const float* e2g = (const float*)d_in[11]; const float* e2b = (const float*)d_in[12];
  const float* e2m = (const float*)d_in[13]; const float* e2v = (const float*)d_in[14];
  const float* enc_wc = (const float*)d_in[15]; const float* enc_bc = (const float*)d_in[16];
  const float* codebook = (const float*)d_in[17];
  const float* dec_w0 = (const float*)d_in[18]; const float* dec_b0 = (const float*)d_in[19];
  const float* dec_w1 = (const float*)d_in[20]; const float* dec_b1 = (const float*)d_in[21];
  const float* d1g = (const float*)d_in[22]; const float* d1b = (const float*)d_in[23];
  const float* d1m = (const float*)d_in[24]; const float* d1v = (const float*)d_in[25];
  const float* dec_w2 = (const float*)d_in[26]; const float* dec_b2 = (const float*)d_in[27];
  const float* d2g = (const float*)d_in[28]; const float* d2b = (const float*)d_in[29];
  const float* d2m = (const float*)d_in[30]; const float* d2v = (const float*)d_in[31];
  const float* dec_wt = (const float*)d_in[32]; const float* dec_bt = (const float*)d_in[33];
  const float* cls_w1 = (const float*)d_in[34]; const float* cls_b1 = (const float*)d_in[35];
  const float* cls_w2 = (const float*)d_in[36]; const float* cls_b2 = (const float*)d_in[37];
  const float* sp_w1 = (const float*)d_in[38]; const float* sp_b1 = (const float*)d_in[39];
  const float* sp_w2 = (const float*)d_in[40]; const float* sp_b2 = (const float*)d_in[41];

  float* out = (float*)d_out;
  const long LOGITS_OFF = 2304000;   // after x_recon (128*9*2*1000)
  const long IDX_OFF    = 2305540;   // after logits + 4 scalars
  const long STATES_OFF = 2433540;   // after indices (128*1000)

  // bump allocator over workspace
  char* ws = (char*)d_ws;
  size_t off = 0;
  auto alloc = [&](size_t bytes) -> void* {
    void* p = ws + off;
    off += (bytes + 255) & ~(size_t)255;
    return p;
  };
  // f16 ping-pong activation buffers: 2 x 65.5MB + 16.4MB -> fits global L2
  h16* bufA = (h16*)alloc((size_t)B * 256 * T * 2);
  h16* bufB = (h16*)alloc((size_t)B * 256 * T * 2);
  h16* bufC = (h16*)alloc((size_t)B * 64 * T * 2);   // z_q
  int*   idxb = (int*)alloc((size_t)B * T * 4);
  float* counts = (float*)alloc(256 * 4);
  float* eacc   = (float*)alloc(4);
  h16* wpe0 = (h16*)alloc(64  * 160 * 2);
  h16* wpe1 = (h16*)alloc(128 * 192 * 2);
  h16* wpe2 = (h16*)alloc(256 * 384 * 2);
  h16* wpec = (h16*)alloc(64  * 256 * 2);
  h16* wpd0 = (h16*)alloc(256 * 64  * 2);
  h16* wpd1 = (h16*)alloc(128 * 768 * 2);
  h16* wpd2 = (h16*)alloc(64  * 384 * 2);
  h16* wpt0 = (h16*)alloc(32 * 320 * 2);
  h16* wpt1 = (h16*)alloc(32 * 320 * 2);
  h16* wps1 = (h16*)alloc(32 * 64 * 2);
  h16* wps2 = (h16*)alloc(32 * 32 * 2);
  h16* cbh    = (h16*)alloc(256 * 64 * 2);
  float* cnorm = (float*)alloc(256 * 4);
  auto abpair = [&](int mpad, float*& a, float*& bt) {
    a = (float*)alloc(mpad * 4); bt = (float*)alloc(mpad * 4);
  };
  float *ae0,*be0,*ae1,*be1,*ae2,*be2,*aec,*bec,*ad0,*bd0,*ad1,*bd1,*ad2,*bd2,*adt,*bdt,*as1,*bs1,*as2,*bs2;
  abpair(64, ae0, be0);  abpair(128, ae1, be1); abpair(256, ae2, be2);
  abpair(64, aec, bec);  abpair(256, ad0, bd0); abpair(128, ad1, bd1);
  abpair(64, ad2, bd2);  abpair(32, adt, bdt);  abpair(32, as1, bs1);
  abpair(32, as2, bs2);

  dim3 blk(256);
  auto gprep = [](long total) { return dim3((unsigned)((total + 255) / 256)); };

  // ---- prep: pack weights to f16, fold BN ----
  prep_w<<<gprep(64L*160),  blk, 0, stream>>>(enc_w0, wpe0,  64,   9, 2, 5,  32, 160,  64);
  prep_w<<<gprep(128L*192), blk, 0, stream>>>(enc_w1, wpe1, 128,  64, 1, 3,  64, 192, 128);
  prep_w<<<gprep(256L*384), blk, 0, stream>>>(enc_w2, wpe2, 256, 128, 1, 3, 128, 384, 256);
  prep_w<<<gprep(64L*256),  blk, 0, stream>>>(enc_wc, wpec,  64, 256, 1, 1, 256, 256,  64);
  prep_w<<<gprep(256L*64),  blk, 0, stream>>>(dec_w0, wpd0, 256,  64, 1, 1,  64,  64, 256);
  prep_w<<<gprep(128L*768), blk, 0, stream>>>(dec_w1, wpd1, 128, 256, 1, 3, 256, 768, 128);
  prep_w<<<gprep(64L*384),  blk, 0, stream>>>(dec_w2, wpd2,  64, 128, 1, 3, 128, 384,  64);
  prep_wt<<<gprep(32L*320), blk, 0, stream>>>(dec_wt, wpt0, wpt1);
  prep_w<<<gprep(32L*64),   blk, 0, stream>>>(sp_w1, wps1,  32,  64, 1, 1,  64,  64,  32);
  prep_w<<<gprep(32L*32),   blk, 0, stream>>>(sp_w2, wps2,  16,  32, 1, 1,  32,  32,  32);
  prep_cb<<<gprep(256L*64), blk, 0, stream>>>(codebook, cbh, cnorm);
  prep_ab<<<1, 256, 0, stream>>>(enc_b0, nullptr, nullptr, nullptr, nullptr, ae0, be0, 64, 64);
  prep_ab<<<1, 256, 0, stream>>>(enc_b1, e1g, e1b, e1m, e1v, ae1, be1, 128, 128);
  prep_ab<<<1, 256, 0, stream>>>(enc_b2, e2g, e2b, e2m, e2v, ae2, be2, 256, 256);
  prep_ab<<<1, 256, 0, stream>>>(enc_bc, nullptr, nullptr, nullptr, nullptr, aec, bec, 64, 64);
  prep_ab<<<1, 256, 0, stream>>>(dec_b0, nullptr, nullptr, nullptr, nullptr, ad0, bd0, 256, 256);
  prep_ab<<<1, 256, 0, stream>>>(dec_b1, d1g, d1b, d1m, d1v, ad1, bd1, 128, 128);
  prep_ab<<<1, 256, 0, stream>>>(dec_b2, d2g, d2b, d2m, d2v, ad2, bd2, 64, 64);
  prep_ab<<<1, 256, 0, stream>>>(dec_bt, nullptr, nullptr, nullptr, nullptr, adt, bdt, 9, 32);
  prep_ab<<<1, 256, 0, stream>>>(sp_b1, nullptr, nullptr, nullptr, nullptr, as1, bs1, 32, 32);
  prep_ab<<<1, 256, 0, stream>>>(sp_b2, nullptr, nullptr, nullptr, nullptr, as2, bs2, 16, 32);

  auto cgrid = [&](int mpad) { return dim3((T + 63) / 64, mpad / 32, B); };

  // ---- encoder ----
  conv_wmma<float, h16,  18,  32, 5, 2,  64, false><<<cgrid(64),  blk, 0, stream>>>(x,    bufA, wpe0, ae0, be0, T, 18L*T,  64L*T, T);
  conv_wmma<h16,   h16,  64,  64, 3, 1, 128, true ><<<cgrid(128), blk, 0, stream>>>(bufA, bufB, wpe1, ae1, be1, T, 64L*T, 128L*T, T);
  conv_wmma<h16,   h16, 128, 128, 3, 1, 256, true ><<<cgrid(256), blk, 0, stream>>>(bufB, bufA, wpe2, ae2, be2, T, 128L*T, 256L*T, T);
  conv_wmma<h16,   h16, 256, 256, 1, 0,  64, false><<<cgrid(64),  blk, 0, stream>>>(bufA, bufB, wpec, aec, bec, T, 256L*T,  64L*T, T);

  // ---- VQ ----
  zero_small<<<1, 256, 0, stream>>>(counts, eacc);
  vq_argmin<<<dim3((T + 127) / 128, 1, B), blk, 0, stream>>>(bufB, cbh, cnorm, idxb, T);
  vq_post<<<dim3((B * T + 255) / 256), blk, 0, stream>>>(idxb, codebook, bufB, bufC,
                                                         out + IDX_OFF, eacc, counts, T);
  vq_finalize<<<1, 256, 0, stream>>>(counts, eacc, out);

  // ---- decoder ----
  conv_wmma<h16, h16,  64,  64, 1, 0, 256, false><<<cgrid(256), blk, 0, stream>>>(bufC, bufA, wpd0, ad0, bd0, T, 64L*T, 256L*T, T);
  conv_wmma<h16, h16, 256, 256, 3, 1, 128, true ><<<cgrid(128), blk, 0, stream>>>(bufA, bufB, wpd1, ad1, bd1, T, 256L*T, 128L*T, T);
  conv_wmma<h16, h16, 128, 128, 3, 1,  64, true ><<<cgrid(64),  blk, 0, stream>>>(bufB, bufA, wpd2, ad2, bd2, T, 128L*T,  64L*T, T);
  // ConvTranspose -> two KH=1 conv launches (one per output H row), f32 out
  conv_wmma<h16, float, 64, 64, 5, 2,  9, false><<<cgrid(32),  blk, 0, stream>>>(bufA, out,     wpt0, adt, bdt, T, 64L*T, 18L*T, 2L*T);
  conv_wmma<h16, float, 64, 64, 5, 2,  9, false><<<cgrid(32),  blk, 0, stream>>>(bufA, out + T, wpt1, adt, bdt, T, 64L*T, 18L*T, 2L*T);

  // ---- classifier + state predictor ----
  cls_kernel<<<dim3(B), blk, 0, stream>>>(bufC, cls_w1, cls_b1, cls_w2, cls_b2,
                                          out + LOGITS_OFF, T);
  conv_wmma<h16, h16,   64, 64, 1, 0, 32, true ><<<cgrid(32), blk, 0, stream>>>(bufC, bufB, wps1, as1, bs1, T, 64L*T, 32L*T, T);
  conv_wmma<h16, float, 32, 32, 1, 0, 16, false><<<cgrid(32), blk, 0, stream>>>(bufB, out + STATES_OFF, wps2, as2, bs2, T, 32L*T, 16L*T, T);
}
